// GraphAttentionNetwork_82952998355941
// MI455X (gfx1250) — matrix operand order repaired and checked by reading, not compiled, not run on hardware
//
#include <hip/hip_runtime.h>

// ---------------------------------------------------------------------------
// 3-layer GAT for MI455X (gfx1250, wave32).
//  - GEMMs: split-precision BF16x3 WMMA (v_wmma_f32_16x16x32_bf16), f32 accum.
//    Each wave computes a 16x64 strip (4 accumulators) to reuse A fragments.
//  - Edge softmax: ordered-uint atomicMax + atomicAdd(f32).
//  - Aggregation: one wave per edge, float4 gathers, global_atomic_add_f32.
// ---------------------------------------------------------------------------

typedef __attribute__((ext_vector_type(16))) __bf16 v16bf;
typedef __attribute__((ext_vector_type(8)))  float  v8f;

#define NEG_SLOPE 0.2f

static __device__ __forceinline__ unsigned short f2bf(float f) {
  unsigned u = __float_as_uint(f);
  u += 0x7FFFu + ((u >> 16) & 1u);          // round-to-nearest-even
  return (unsigned short)(u >> 16);
}
static __device__ __forceinline__ float bf2f(unsigned short b) {
  return __uint_as_float(((unsigned)b) << 16);
}
// total-order encode for float atomicMax via u32
static __device__ __forceinline__ unsigned ford(float f) {
  unsigned b = __float_as_uint(f);
  return (b & 0x80000000u) ? ~b : (b | 0x80000000u);
}
static __device__ __forceinline__ float finv(unsigned o) {
  return __uint_as_float((o & 0x80000000u) ? (o ^ 0x80000000u) : ~o);
}

// ---------------- conversion kernels ----------------
__global__ void k_convert_hilo(const float* __restrict__ X,
                               unsigned short* __restrict__ hi,
                               unsigned short* __restrict__ lo, int n) {
  int i = blockIdx.x * blockDim.x + threadIdx.x;
  if (i >= n) return;
  float x = X[i];
  unsigned short h = f2bf(x);
  hi[i] = h;
  lo[i] = f2bf(x - bf2f(h));
}

// W[K,Nc] f32 -> Wt_hi/Wt_lo[Nc,K] bf16 (transposed so B-columns are contiguous)
__global__ void k_convert_w(const float* __restrict__ W,
                            unsigned short* __restrict__ whi,
                            unsigned short* __restrict__ wlo, int K, int Nc) {
  int i = blockIdx.x * blockDim.x + threadIdx.x;
  if (i >= K * Nc) return;
  int k = i / Nc, n = i % Nc;
  float x = W[i];
  unsigned short h = f2bf(x);
  whi[(size_t)n * K + k] = h;
  wlo[(size_t)n * K + k] = f2bf(x - bf2f(h));
}

__global__ void k_zero(float* __restrict__ p, int n) {
  int i = blockIdx.x * blockDim.x + threadIdx.x;
  if (i < n) p[i] = 0.0f;
}

// ---------------- WMMA GEMM: Y[M,Nc] = X[M,K] @ W[K,Nc], bf16x3 ----------------
// A fragment per ISA 7.12.2: lane L holds row (L&15); K chunks of 8 at
// (L>>4)*8 and 16+(L>>4)*8.  B: lane L holds column (L&15), same K chunks
// as A (weights pre-transposed to [Nc,K]).
static __device__ __forceinline__ v16bf load_a(const unsigned short* __restrict__ X,
                                               int row, int K, int k0, int lane) {
  const unsigned short* p = X + (size_t)row * K + k0 + ((lane >> 4) << 3);
  union { uint4 u[2]; v16bf v; } t;
  t.u[0] = *(const uint4*)p;         // K = kb+0..7
  t.u[1] = *(const uint4*)(p + 16);  // K = kb+16..23
  return t.v;
}
static __device__ __forceinline__ v16bf load_b(const unsigned short* __restrict__ Wt,
                                               int col, int K, int k0, int lane) {
  const unsigned short* p = Wt + (size_t)col * K + k0 + ((lane >> 4) << 3);
  union { uint4 u[2]; v16bf v; } t;
  t.u[0] = *(const uint4*)p;        // K = kb+0..7
  t.u[1] = *(const uint4*)(p + 16); // K = kb+16..23
  return t.v;
}

// one wave -> 16x64 output strip (4 column tiles), A fragments reused 4x
__global__ void k_gemm_bf16x3(const unsigned short* __restrict__ Xhi,
                              const unsigned short* __restrict__ Xlo,
                              const unsigned short* __restrict__ Whi,
                              const unsigned short* __restrict__ Wlo,
                              float* __restrict__ Y, int M, int K, int Nc) {
  int wave = (blockIdx.x * blockDim.x + threadIdx.x) >> 5;
  int lane = threadIdx.x & 31;
  int groupsN = Nc >> 6;                     // 64-wide column groups
  int strips = (M >> 4) * groupsN;
  if (wave >= strips) return;                // wave-uniform: EXEC stays full
  int tm = wave / groupsN, tg = wave % groupsN;
  int row  = tm * 16 + (lane & 15);
  int col0 = tg * 64 + (lane & 15);

  v8f acc[4];
#pragma unroll
  for (int t = 0; t < 4; ++t) acc[t] = (v8f){0.f,0.f,0.f,0.f,0.f,0.f,0.f,0.f};

  for (int k0 = 0; k0 < K; k0 += 32) {
    v16bf ah = load_a(Xhi, row, K, k0, lane);
    v16bf al = load_a(Xlo, row, K, k0, lane);
#pragma unroll
    for (int t = 0; t < 4; ++t) {
      v16bf bh = load_b(Whi, col0 + t * 16, K, k0, lane);
      v16bf bl = load_b(Wlo, col0 + t * 16, K, k0, lane);
      acc[t] = __builtin_amdgcn_wmma_f32_16x16x32_bf16(false, ah, false, bh, (short)0, acc[t], false, false);
      acc[t] = __builtin_amdgcn_wmma_f32_16x16x32_bf16(false, ah, false, bl, (short)0, acc[t], false, false);
      acc[t] = __builtin_amdgcn_wmma_f32_16x16x32_bf16(false, al, false, bh, (short)0, acc[t], false, false);
    }
  }
  // D layout: VGPR r -> row r + 8*(lane>>4), col = lane&15
  float* ybase = Y + (size_t)(tm * 16 + (lane >> 4) * 8) * Nc;
#pragma unroll
  for (int t = 0; t < 4; ++t) {
    float* yp = ybase + col0 + t * 16;
#pragma unroll
    for (int r = 0; r < 8; ++r) yp[(size_t)r * Nc] = acc[t][r];
  }
}

// ---------------- per-node attention half-logits ----------------
__global__ void k_node_logits(const float* __restrict__ Hf,
                              const float* __restrict__ asrc,
                              const float* __restrict__ adst,
                              float* __restrict__ als, float* __restrict__ ald,
                              int Nn, int H, int C) {
  int i = blockIdx.x * blockDim.x + threadIdx.x;
  if (i >= Nn * H) return;
  int n = i / H, h = i % H;
  const float* hp = Hf + (size_t)n * H * C + h * C;
  const float* as = asrc + h * C;
  const float* ad = adst + h * C;
  float s = 0.f, d = 0.f;
  for (int c = 0; c < C; c += 4) {
    float4 v = *(const float4*)(hp + c);
    float4 a = *(const float4*)(as + c);
    float4 b = *(const float4*)(ad + c);
    s += v.x * a.x + v.y * a.y + v.z * a.z + v.w * a.w;
    d += v.x * b.x + v.y * b.y + v.z * b.z + v.w * b.w;
  }
  als[i] = s;
  ald[i] = d;
}

// ---------------- edge passes ----------------
__global__ void k_edge_max(const int* __restrict__ ei, int E,
                           const float* __restrict__ als, const float* __restrict__ ald,
                           float* __restrict__ eval, unsigned* __restrict__ emax,
                           int H, int Etot) {
  int e = blockIdx.x * blockDim.x + threadIdx.x;
  if (e >= Etot) return;
  int s = (e < E) ? ei[e] : (e - E);
  int d = (e < E) ? ei[E + e] : (e - E);
  for (int h = 0; h < H; ++h) {
    float v = als[s * H + h] + ald[d * H + h];
    v = (v > 0.f) ? v : NEG_SLOPE * v;          // leaky_relu
    eval[(size_t)e * H + h] = v;
    atomicMax(&emax[d * H + h], ford(v));
  }
}

__global__ void k_edge_expsum(const int* __restrict__ ei, int E,
                              float* __restrict__ eval,
                              const unsigned* __restrict__ emax,
                              float* __restrict__ den, int H, int Etot) {
  int e = blockIdx.x * blockDim.x + threadIdx.x;
  if (e >= Etot) return;
  int d = (e < E) ? ei[E + e] : (e - E);
  for (int h = 0; h < H; ++h) {
    float m = finv(emax[d * H + h]);
    float ex = __expf(eval[(size_t)e * H + h] - m);
    eval[(size_t)e * H + h] = ex;
    atomicAdd(&den[d * H + h], ex);
  }
}

// wave per edge; lane covers HC/32 consecutive channels (all in one head)
__global__ void k_aggregate(const int* __restrict__ ei, int E,
                            const float* __restrict__ Hf,
                            const float* __restrict__ eval,
                            const float* __restrict__ den,
                            float* __restrict__ out,
                            int H, int C, int HCl, int Etot) {
  int gid = blockIdx.x * blockDim.x + threadIdx.x;
  int e = gid >> 5;
  int lane = gid & 31;
  if (e >= Etot) return;
  int s = (e < E) ? ei[e] : (e - E);
  int d = (e < E) ? ei[E + e] : (e - E);
  int cpl = HCl >> 5;                 // channels per lane (16 or 4)
  int cb = lane * cpl;
  int h = cb / C;                     // head is lane-uniform (cpl divides C)
  float alpha = eval[(size_t)e * H + h] / (den[d * H + h] + 1e-16f);
  const float* hs = Hf + (size_t)s * HCl;
  float* od = out + (size_t)d * HCl;
  for (int c = cb; c < cb + cpl; c += 4) {
    float4 v = *(const float4*)(hs + c);
    atomicAdd(od + c + 0, alpha * v.x);
    atomicAdd(od + c + 1, alpha * v.y);
    atomicAdd(od + c + 2, alpha * v.z);
    atomicAdd(od + c + 3, alpha * v.w);
  }
}

__global__ void k_bias_act(const float* __restrict__ agg, const float* __restrict__ b,
                           float* __restrict__ out, int Nn, int HCl, int do_elu) {
  int i = blockIdx.x * blockDim.x + threadIdx.x;
  if (i >= Nn * HCl) return;
  float v = agg[i] + b[i % HCl];
  if (do_elu) v = (v > 0.f) ? v : (__expf(v) - 1.0f);
  out[i] = v;
}

// ---------------------------------------------------------------------------
static inline int cdiv(long long a, int b) { return (int)((a + b - 1) / b); }

extern "C" void kernel_launch(void* const* d_in, const int* in_sizes, int n_in,
                              void* d_out, int out_size, void* d_ws, size_t ws_size,
                              hipStream_t stream) {
  const float* x    = (const float*)d_in[0];
  const int*   ei   = (const int*)  d_in[1];
  const float* W1   = (const float*)d_in[2];
  const float* as1  = (const float*)d_in[3];
  const float* ad1  = (const float*)d_in[4];
  const float* b1   = (const float*)d_in[5];
  const float* W2   = (const float*)d_in[6];
  const float* as2  = (const float*)d_in[7];
  const float* ad2  = (const float*)d_in[8];
  const float* b2   = (const float*)d_in[9];
  const float* W3   = (const float*)d_in[10];
  const float* as3  = (const float*)d_in[11];
  const float* ad3  = (const float*)d_in[12];
  const float* b3   = (const float*)d_in[13];

  const int Nn   = in_sizes[0] / 128;   // 20000
  const int E    = in_sizes[1] / 2;     // 320000
  const int Etot = E + Nn;              // + self loops

  // workspace carve-up (256B aligned)
  char* w = (char*)d_ws;
  size_t off = 0;
  auto carve = [&](size_t bytes) -> void* {
    void* p = w + off;
    off = (off + bytes + 255) & ~(size_t)255;
    return p;
  };
  float*          hbuf  = (float*)         carve((size_t)Nn * 512 * 4);
  float*          agg   = (float*)         carve((size_t)Nn * 512 * 4);
  unsigned short* Xhi   = (unsigned short*)carve((size_t)Nn * 512 * 2);
  unsigned short* Xlo   = (unsigned short*)carve((size_t)Nn * 512 * 2);
  unsigned short* Whi   = (unsigned short*)carve((size_t)512 * 512 * 2);
  unsigned short* Wlo   = (unsigned short*)carve((size_t)512 * 512 * 2);
  float*          als   = (float*)         carve((size_t)Nn * 8 * 4);
  float*          ald   = (float*)         carve((size_t)Nn * 8 * 4);
  unsigned*       emax  = (unsigned*)      carve((size_t)Nn * 8 * 4);
  float*          den   = (float*)         carve((size_t)Nn * 8 * 4);
  float*          eval  = (float*)         carve((size_t)Etot * 8 * 4);
  (void)ws_size; (void)n_in; (void)out_size;

  auto layer = [&](const float* feat_in, int K, const float* W, int Nc,
                   const float* asrc, const float* adst, const float* bias,
                   int H, int C, float* feat_out, int do_elu) {
    int nIn = Nn * K;
    k_convert_hilo<<<cdiv(nIn, 256), 256, 0, stream>>>(feat_in, Xhi, Xlo, nIn);
    k_convert_w<<<cdiv((long long)K * Nc, 256), 256, 0, stream>>>(W, Whi, Wlo, K, Nc);
    int strips = (Nn / 16) * (Nc / 64);   // one wave per 16x64 strip
    k_gemm_bf16x3<<<cdiv((long long)strips * 32, 128), 128, 0, stream>>>(
        Xhi, Xlo, Whi, Wlo, hbuf, Nn, K, Nc);
    k_node_logits<<<cdiv((long long)Nn * H, 256), 256, 0, stream>>>(
        hbuf, asrc, adst, als, ald, Nn, H, C);
    int HCl = H * C;
    k_zero<<<cdiv((long long)Nn * H, 256), 256, 0, stream>>>((float*)emax, Nn * H);
    k_zero<<<cdiv((long long)Nn * H, 256), 256, 0, stream>>>(den, Nn * H);
    k_zero<<<cdiv((long long)Nn * HCl, 256), 256, 0, stream>>>(agg, Nn * HCl);
    k_edge_max<<<cdiv(Etot, 256), 256, 0, stream>>>(ei, E, als, ald, eval, emax, H, Etot);
    k_edge_expsum<<<cdiv(Etot, 256), 256, 0, stream>>>(ei, E, eval, emax, den, H, Etot);
    k_aggregate<<<cdiv((long long)Etot * 32, 256), 256, 0, stream>>>(
        ei, E, hbuf, eval, den, agg, H, C, HCl, Etot);
    k_bias_act<<<cdiv((long long)Nn * HCl, 256), 256, 0, stream>>>(
        agg, bias, feat_out, Nn, HCl, do_elu);
  };

  // layer 1: x[N,128] -> hbuf[N,512], ELU   (feat_out overwrites hbuf after agg)
  layer(x,    128, W1, 512, as1, ad1, b1, 8, 64,  hbuf, 1);
  // layer 2: hbuf[N,512] -> hbuf[N,512], ELU (hbuf converted to bf16 first)
  layer(hbuf, 512, W2, 512, as2, ad2, b2, 8, 64,  hbuf, 1);
  // layer 3: hbuf[N,512] -> d_out[N,128], heads=1, no ELU
  layer(hbuf, 512, W3, 128, as3, ad3, b3, 1, 128, (float*)d_out, 0);
}
